// GNNAgent_39719857554100
// MI455X (gfx1250) — matrix operand order repaired and checked by reading, not compiled, not run on hardware
//
#include <hip/hip_runtime.h>
#include <hip/hip_bf16.h>

// ---------------------------------------------------------------------------
// GNNAgent forward for MI455X (gfx1250, wave32, WMMA bf16)
// ---------------------------------------------------------------------------

#define N_NODES 65536
#define N_EDGES 1048576
#define IN_DIM  96
#define HID     128
#define OUT_DIM 256
#define N_ACT   20
#define NEG_SLOPE 0.2f
#define LN_EPS  1e-5f

typedef __bf16 bf16_t;
typedef bf16_t v16bf  __attribute__((ext_vector_type(16)));
typedef bf16_t bf16x8 __attribute__((ext_vector_type(8)));
typedef float  v8f    __attribute__((ext_vector_type(8)));

union V16U { v16bf v; bf16x8 h[2]; };

// A-fragment (16x32 bf16, wave32): lane holds row M = lane&15.
// K layout per lane: contiguous [kbase, kbase+8) in elements 0..7 and
// [kbase+16, kbase+24) in elements 8..15, kbase = ktile*32 + 8*(lane>=16).
__device__ __forceinline__ v16bf load_a_frag(const bf16_t* rowbase, int kbase) {
    V16U u;
    u.h[0] = *(const bf16x8*)(rowbase + kbase);
    u.h[1] = *(const bf16x8*)(rowbase + kbase + 16);
    return u.v;
}

// B-fragment from pre-packed weights: each lane reads 16 contiguous bf16 (32B).
__device__ __forceinline__ v16bf load_b_frag(const bf16_t* p) {
    V16U u;
    u.h[0] = *(const bf16x8*)(p);
    u.h[1] = *(const bf16x8*)(p + 8);
    return u.v;
}

__device__ __forceinline__ v8f wmma_bf16(v16bf a, v16bf b, v8f c) {
    return __builtin_amdgcn_wmma_f32_16x16x32_bf16(false, a, false, b,
                                                   (short)0, c, false, false);
}

// sign-aware float atomic max (init slot must be -inf)
__device__ __forceinline__ void atomicMaxF(float* addr, float val) {
    if (val >= 0.f) atomicMax((int*)addr, __float_as_int(val));
    else            atomicMin((unsigned int*)addr, __float_as_uint(val));
}

// ---------------------------------------------------------------------------
// Pack f32 weight [K][Nc] -> bf16 WMMA-B fragment layout, tile-major.
// tileIdx = kt*(NcPad/16)+nt ; within tile: lane-major, 16 elems/lane.
// element e of lane L in tile (kt,nt): K = kt*32 + 8*(L>=16) + (e&7) + 16*(e>=8),
//                                      N = nt*16 + (L&15). Pad N>=Nc with 0.
// ---------------------------------------------------------------------------
__global__ void pack_b_kernel(const float* __restrict__ W, bf16_t* __restrict__ dst,
                              int K, int Nc, int NcPad) {
    int t = blockIdx.x * 256 + threadIdx.x;
    int total = K * NcPad;
    if (t >= total) return;
    int nNT  = NcPad >> 4;
    int tile = t >> 9;
    int r    = t & 511;
    int lane = r >> 4;
    int e    = r & 15;
    int kt = tile / nNT, nt = tile % nNT;
    int k = kt * 32 + ((lane >> 4) << 3) + (e & 7) + ((e >> 3) << 4);
    int n = nt * 16 + (lane & 15);
    float v = (n < Nc) ? W[k * Nc + n] : 0.f;
    dst[t] = (bf16_t)v;
}

__global__ void init_node_kernel(float* __restrict__ nmax, float* __restrict__ denom) {
    int i = blockIdx.x * 256 + threadIdx.x;
    nmax[i]  = -INFINITY;
    denom[i] = 0.f;
}

// ---------------------------------------------------------------------------
// MLP layer: out = LN(ReLU(X @ W + b)) ; 16 rows/block, 8 waves = 128 cols.
// ---------------------------------------------------------------------------
template<int KDIM, bool IN_F32>
__global__ __launch_bounds__(256)
void mlp_layer_kernel(const void* __restrict__ xin_, const bf16_t* __restrict__ Wp,
                      const float* __restrict__ bias, const float* __restrict__ gam,
                      const float* __restrict__ bet, bf16_t* __restrict__ xout) {
    constexpr int KS = KDIM + 8;          // padded LDS stride (bank spread)
    __shared__ bf16_t As[16 * KS];
    __shared__ float  Cs[16][HID];
    __shared__ float  mu_s[16], rs_s[16];

    const int tid = threadIdx.x;
    const size_t row0 = (size_t)blockIdx.x * 16;

    if (IN_F32) {
        const float* xin = (const float*)xin_;
        for (int i = tid; i < 16 * KDIM; i += 256) {
            int r = i / KDIM, k = i % KDIM;
            As[r * KS + k] = (bf16_t)xin[(row0 + r) * KDIM + k];
        }
    } else {
        const bf16_t* xin = (const bf16_t*)xin_;
        for (int i = tid; i < 16 * KDIM; i += 256) {
            int r = i / KDIM, k = i % KDIM;
            As[r * KS + k] = xin[(row0 + r) * KDIM + k];
        }
    }
    __syncthreads();

    const int wave = tid >> 5, lane = tid & 31;
    const int m = lane & 15;
    v8f acc = {0.f, 0.f, 0.f, 0.f, 0.f, 0.f, 0.f, 0.f};
    constexpr int NKT = KDIM / 32;
#pragma unroll
    for (int kt = 0; kt < NKT; ++kt) {
        v16bf a = load_a_frag(&As[m * KS], kt * 32 + ((lane >> 4) << 3));
        v16bf b = load_b_frag(Wp + (((kt << 3) + wave) << 9) + (lane << 4));
        acc = wmma_bf16(a, b, acc);
    }
    const int n = (wave << 4) + m;
    const float bn = bias[n];
    const int mbase = (lane >> 4) << 3;
#pragma unroll
    for (int v = 0; v < 8; ++v) {
        float val = acc[v] + bn;
        Cs[mbase + v][n] = val > 0.f ? val : 0.f;   // ReLU before LN
    }
    __syncthreads();

    if (tid < 16) {
        float s = 0.f, s2 = 0.f;
        for (int c = 0; c < HID; ++c) { float v = Cs[tid][c]; s += v; s2 += v * v; }
        float mu = s * (1.f / HID);
        float var = s2 * (1.f / HID) - mu * mu;
        mu_s[tid] = mu;
        rs_s[tid] = __frsqrt_rn(var + LN_EPS);
    }
    __syncthreads();

    {
        int r = tid >> 4, c0 = (tid & 15) << 3;
        float mu = mu_s[r], rs = rs_s[r];
#pragma unroll
        for (int j = 0; j < 8; ++j) {
            int c = c0 + j;
            float val = (Cs[r][c] - mu) * rs * gam[c] + bet[c];
            xout[(row0 + r) * HID + c] = (bf16_t)val;
        }
    }
}

// ---------------------------------------------------------------------------
// GAT linear transforms: blockIdx.y selects Wl (->xl bf16), Wr (->xr bf16),
// Wres (->h f32 base, includes gat_bias). 16 rows/block; wave does 2 N-tiles.
// ---------------------------------------------------------------------------
__global__ __launch_bounds__(256)
void gat_lin_kernel(const bf16_t* __restrict__ x2,
                    const bf16_t* __restrict__ Wlp, const bf16_t* __restrict__ Wrp,
                    const bf16_t* __restrict__ Wresp,
                    const float* __restrict__ bl, const float* __restrict__ br,
                    const float* __restrict__ gbias,
                    bf16_t* __restrict__ xl, bf16_t* __restrict__ xr,
                    float* __restrict__ h) {
    constexpr int KS = HID + 8;
    __shared__ bf16_t As[16 * KS];

    const int tid = threadIdx.x;
    const size_t row0 = (size_t)blockIdx.x * 16;
    for (int i = tid; i < 16 * HID; i += 256) {
        int r = i / HID, k = i % HID;
        As[r * KS + k] = x2[(row0 + r) * HID + k];
    }
    __syncthreads();

    const int which = blockIdx.y;
    const bf16_t* Wp = (which == 0) ? Wlp : (which == 1) ? Wrp : Wresp;
    const float*  bb = (which == 0) ? bl  : (which == 1) ? br  : gbias;

    const int wave = tid >> 5, lane = tid & 31;
    const int m = lane & 15;
    const int mbase = (lane >> 4) << 3;

#pragma unroll
    for (int half = 0; half < 2; ++half) {
        const int nt = wave + (half << 3);      // 0..15 (nNT = 16)
        v8f acc = {0.f, 0.f, 0.f, 0.f, 0.f, 0.f, 0.f, 0.f};
#pragma unroll
        for (int kt = 0; kt < HID / 32; ++kt) {
            v16bf a = load_a_frag(&As[m * KS], kt * 32 + ((lane >> 4) << 3));
            v16bf b = load_b_frag(Wp + (((kt << 4) + nt) << 9) + (lane << 4));
            acc = wmma_bf16(a, b, acc);
        }
        const int n = (nt << 4) + m;
        const float bn = bb[n];
#pragma unroll
        for (int v = 0; v < 8; ++v) {
            float val = acc[v] + bn;
            size_t idx = (row0 + mbase + v) * OUT_DIM + n;
            if (which == 0)      xl[idx] = (bf16_t)val;
            else if (which == 1) xr[idx] = (bf16_t)val;
            else                 h[idx]  = val;     // residual base + gat_bias
        }
    }
}

// ---------------------------------------------------------------------------
// Edge attention score: wave per edge; e = dot(lrelu(xl[s]+xr[d]+ea@We), att)
// followed by segment max via atomics.
// ---------------------------------------------------------------------------
__global__ __launch_bounds__(256)
void edge_score_kernel(const int* __restrict__ ei, const float* __restrict__ ea,
                       const bf16_t* __restrict__ xl, const bf16_t* __restrict__ xr,
                       const float* __restrict__ We, const float* __restrict__ att,
                       float* __restrict__ e_out, float* __restrict__ nmax) {
    const int gid  = blockIdx.x * 256 + threadIdx.x;
    const int edge = gid >> 5;
    const int lane = gid & 31;
    if (edge >= N_EDGES) return;
    const int src = ei[edge];
    const int dst = ei[N_EDGES + edge];
    const float a0 = ea[edge * 3 + 0];
    const float a1 = ea[edge * 3 + 1];
    const float a2 = ea[edge * 3 + 2];
    const int n0 = lane << 3;
    bf16x8 l8 = *(const bf16x8*)(xl + (size_t)src * OUT_DIM + n0);
    bf16x8 r8 = *(const bf16x8*)(xr + (size_t)dst * OUT_DIM + n0);
    float part = 0.f;
#pragma unroll
    for (int j = 0; j < 8; ++j) {
        int n = n0 + j;
        float mm = (float)l8[j] + (float)r8[j]
                 + a0 * We[n] + a1 * We[OUT_DIM + n] + a2 * We[2 * OUT_DIM + n];
        float lr = mm > 0.f ? mm : NEG_SLOPE * mm;
        part += lr * att[n];
    }
    for (int off = 16; off > 0; off >>= 1) part += __shfl_down(part, off, 32);
    if (lane == 0) {
        e_out[edge] = part;
        atomicMaxF(&nmax[dst], part);
    }
}

// ex = exp(e - max[dst]) ; denom[dst] += ex
__global__ __launch_bounds__(256)
void edge_exp_kernel(const int* __restrict__ ei, const float* __restrict__ e_in,
                     const float* __restrict__ nmax, float* __restrict__ ex_out,
                     float* __restrict__ denom) {
    const int i = blockIdx.x * 256 + threadIdx.x;
    if (i >= N_EDGES) return;
    const int dst = ei[N_EDGES + i];
    const float v = __expf(e_in[i] - nmax[dst]);
    ex_out[i] = v;
    atomicAdd(&denom[dst], v);
}

// h[dst] += (ex/denom[dst]) * xl[src] ; wave per edge, 8 dims per lane
__global__ __launch_bounds__(256)
void edge_aggr_kernel(const int* __restrict__ ei, const float* __restrict__ ex,
                      const float* __restrict__ denom, const bf16_t* __restrict__ xl,
                      float* __restrict__ h) {
    const int gid  = blockIdx.x * 256 + threadIdx.x;
    const int edge = gid >> 5;
    const int lane = gid & 31;
    if (edge >= N_EDGES) return;
    const int src = ei[edge];
    const int dst = ei[N_EDGES + edge];
    const float alpha = ex[edge] / denom[dst];   // denom > 0: dst has this edge
    const int n0 = lane << 3;
    bf16x8 l8 = *(const bf16x8*)(xl + (size_t)src * OUT_DIM + n0);
    float* hd = h + (size_t)dst * OUT_DIM + n0;
#pragma unroll
    for (int j = 0; j < 8; ++j) atomicAdd(&hd[j], alpha * (float)l8[j]);
}

// ---------------------------------------------------------------------------
// Head: q = LN(ReLU(h)) @ Wq + bq  (Wq padded 20 -> 32 cols; 2 WMMA N-tiles)
// ---------------------------------------------------------------------------
__global__ __launch_bounds__(256)
void head_kernel(const float* __restrict__ h, const float* __restrict__ g2,
                 const float* __restrict__ be2, const bf16_t* __restrict__ Wqp,
                 const float* __restrict__ bq, float* __restrict__ q) {
    constexpr int KS = OUT_DIM + 8;
    __shared__ bf16_t As[16 * KS];
    __shared__ float  mu_s[16], rs_s[16];

    const int tid = threadIdx.x;
    const size_t row0 = (size_t)blockIdx.x * 16;

    if (tid < 16) {
        float s = 0.f, s2 = 0.f;
        const float* hr = h + (row0 + tid) * OUT_DIM;
        for (int c = 0; c < OUT_DIM; ++c) {
            float v = hr[c]; v = v > 0.f ? v : 0.f;
            s += v; s2 += v * v;
        }
        float mu = s * (1.f / OUT_DIM);
        float var = s2 * (1.f / OUT_DIM) - mu * mu;
        mu_s[tid] = mu;
        rs_s[tid] = __frsqrt_rn(var + LN_EPS);
    }
    __syncthreads();
    {
        int r = tid >> 4, c0 = (tid & 15) << 4;
        float mu = mu_s[r], rs = rs_s[r];
        const float* hr = h + (row0 + r) * OUT_DIM;
#pragma unroll
        for (int j = 0; j < 16; ++j) {
            int c = c0 + j;
            float v = hr[c]; v = v > 0.f ? v : 0.f;
            As[r * KS + c] = (bf16_t)((v - mu) * rs * g2[c] + be2[c]);
        }
    }
    __syncthreads();

    const int wave = tid >> 5, lane = tid & 31;
    const int m = lane & 15;
    if (wave < 2) {                               // 32-col output = 2 N-tiles
        v8f acc = {0.f, 0.f, 0.f, 0.f, 0.f, 0.f, 0.f, 0.f};
#pragma unroll
        for (int kt = 0; kt < OUT_DIM / 32; ++kt) {
            v16bf a = load_a_frag(&As[m * KS], kt * 32 + ((lane >> 4) << 3));
            v16bf b = load_b_frag(Wqp + (((kt << 1) + wave) << 9) + (lane << 4));
            acc = wmma_bf16(a, b, acc);
        }
        const int n = (wave << 4) + m;
        if (n < N_ACT) {
            const float bn = bq[n];
            const int mbase = (lane >> 4) << 3;
#pragma unroll
            for (int v = 0; v < 8; ++v)
                q[(row0 + mbase + v) * N_ACT + n] = acc[v] + bn;
        }
    }
}

// ---------------------------------------------------------------------------
extern "C" void kernel_launch(void* const* d_in, const int* in_sizes, int n_in,
                              void* d_out, int out_size, void* d_ws, size_t ws_size,
                              hipStream_t stream) {
    const float* inputs = (const float*)d_in[0];
    const int*   ei     = (const int*)  d_in[1];
    const float* ea     = (const float*)d_in[2];
    const float* W0     = (const float*)d_in[3];
    const float* b0     = (const float*)d_in[4];
    const float* g0     = (const float*)d_in[5];
    const float* be0    = (const float*)d_in[6];
    const float* W1     = (const float*)d_in[7];
    const float* b1     = (const float*)d_in[8];
    const float* g1     = (const float*)d_in[9];
    const float* be1    = (const float*)d_in[10];
    const float* Wl     = (const float*)d_in[11];
    const float* bl     = (const float*)d_in[12];
    const float* Wr     = (const float*)d_in[13];
    const float* br     = (const float*)d_in[14];
    const float* We     = (const float*)d_in[15];
    const float* att    = (const float*)d_in[16];
    const float* Wres   = (const float*)d_in[17];
    const float* gbias  = (const float*)d_in[18];
    const float* g2     = (const float*)d_in[19];
    const float* be2    = (const float*)d_in[20];
    const float* Wq     = (const float*)d_in[21];
    const float* bq     = (const float*)d_in[22];
    float* q = (float*)d_out;

    char* ws = (char*)d_ws;
    size_t off = 0;
    auto take = [&](size_t bytes) -> char* {
        char* p = ws + off;
        off += (bytes + 255) & ~(size_t)255;
        return p;
    };
    bf16_t* W0p   = (bf16_t*)take((size_t)IN_DIM * HID * 2);
    bf16_t* W1p   = (bf16_t*)take((size_t)HID * HID * 2);
    bf16_t* Wlp   = (bf16_t*)take((size_t)HID * OUT_DIM * 2);
    bf16_t* Wrp   = (bf16_t*)take((size_t)HID * OUT_DIM * 2);
    bf16_t* Wresp = (bf16_t*)take((size_t)HID * OUT_DIM * 2);
    bf16_t* Wqp   = (bf16_t*)take((size_t)OUT_DIM * 32 * 2);
    bf16_t* x1    = (bf16_t*)take((size_t)N_NODES * HID * 2);
    bf16_t* x2    = (bf16_t*)take((size_t)N_NODES * HID * 2);
    bf16_t* xl    = (bf16_t*)take((size_t)N_NODES * OUT_DIM * 2);   // 32 MB: L2-resident
    bf16_t* xr    = (bf16_t*)take((size_t)N_NODES * OUT_DIM * 2);   // 32 MB: L2-resident
    float*  hbuf  = (float*) take((size_t)N_NODES * OUT_DIM * 4);
    float*  e_buf = (float*) take((size_t)N_EDGES * 4);
    float*  exbuf = (float*) take((size_t)N_EDGES * 4);
    float*  nmax  = (float*) take((size_t)N_NODES * 4);
    float*  denom = (float*) take((size_t)N_NODES * 4);
    (void)ws_size; (void)in_sizes; (void)n_in; (void)out_size;

    // 1. pack weights to WMMA-B layout (bf16)
    pack_b_kernel<<<(IN_DIM * HID + 255) / 256, 256, 0, stream>>>(W0, W0p, IN_DIM, HID, HID);
    pack_b_kernel<<<(HID * HID + 255) / 256, 256, 0, stream>>>(W1, W1p, HID, HID, HID);
    pack_b_kernel<<<(HID * OUT_DIM + 255) / 256, 256, 0, stream>>>(Wl, Wlp, HID, OUT_DIM, OUT_DIM);
    pack_b_kernel<<<(HID * OUT_DIM + 255) / 256, 256, 0, stream>>>(Wr, Wrp, HID, OUT_DIM, OUT_DIM);
    pack_b_kernel<<<(HID * OUT_DIM + 255) / 256, 256, 0, stream>>>(Wres, Wresp, HID, OUT_DIM, OUT_DIM);
    pack_b_kernel<<<(OUT_DIM * 32 + 255) / 256, 256, 0, stream>>>(Wq, Wqp, OUT_DIM, N_ACT, 32);

    // 2. init segment-softmax state
    init_node_kernel<<<N_NODES / 256, 256, 0, stream>>>(nmax, denom);

    // 3. node MLP (WMMA + fused ReLU/LN)
    mlp_layer_kernel<IN_DIM, true ><<<N_NODES / 16, 256, 0, stream>>>(inputs, W0p, b0, g0, be0, x1);
    mlp_layer_kernel<HID,    false><<<N_NODES / 16, 256, 0, stream>>>(x1,     W1p, b1, g1, be1, x2);

    // 4. GAT transforms: xl, xr (bf16, L2-resident gather tables), h base
    gat_lin_kernel<<<dim3(N_NODES / 16, 3), 256, 0, stream>>>(
        x2, Wlp, Wrp, Wresp, bl, br, gbias, xl, xr, hbuf);

    // 5. edge attention: score + segment-max, exp + segment-sum, aggregate
    edge_score_kernel<<<(N_EDGES * 32) / 256, 256, 0, stream>>>(ei, ea, xl, xr, We, att, e_buf, nmax);
    edge_exp_kernel<<<N_EDGES / 256, 256, 0, stream>>>(ei, e_buf, nmax, exbuf, denom);
    edge_aggr_kernel<<<(N_EDGES * 32) / 256, 256, 0, stream>>>(ei, exbuf, denom, xl, hbuf);

    // 6. head: LN(ReLU(h)) @ Wq + bq
    head_kernel<<<N_NODES / 16, 256, 0, stream>>>(hbuf, g2, be2, Wqp, bq, q);
}
